// MoEWithTransformer_84318797955294
// MI455X (gfx1250) — matrix operand order repaired and checked by reading, not compiled, not run on hardware
//
#include <hip/hip_runtime.h>

// ---------------- model constants (match reference) ----------------
#define DM    1024   // d_model
#define NHH   16     // heads
#define DHD   64     // head dim
#define NLAY  2
#define DFFN  4096
#define HIDE  4096
#define NE    8      // experts
#define VTGT  32000
#define BBS   2      // batch
#define SSRC  1024
#define STGT  512

typedef __attribute__((ext_vector_type(16))) __bf16 v16bf;
typedef __attribute__((ext_vector_type(8)))  float  v8f;

__device__ __forceinline__ unsigned short f2bf(float f) {
  unsigned u = __float_as_uint(f);
  u += 0x7FFFu + ((u >> 16) & 1u);          // round-to-nearest-even
  return (unsigned short)(u >> 16);
}
__device__ __forceinline__ unsigned pack2(float lo, float hi) {
  return (unsigned)f2bf(lo) | ((unsigned)f2bf(hi) << 16);
}

// LDS tile: row stride 40 ushorts (80 B): 16B-aligned rows, and the 16
// consecutive rows touched by one fragment read map to distinct banks.
#define LDK 40

// build the two A fragments and four B fragments, issue 8 WMMAs
__device__ __forceinline__ void wmma_tile(const unsigned short (*Asb)[LDK],
                                          const unsigned short (*Bsb)[LDK],
                                          int wm, int wn, int lane,
                                          v8f (&acc)[2][4]) {
  union UF { v16bf v; unsigned u[8]; };
  UF afr[2], bfr[4];
  const int kh = (lane >> 4) * 8;          // A: lanes 16..31 hold K+8 group
#pragma unroll
  for (int i = 0; i < 2; ++i) {
    const int r = wm * 32 + i * 16 + (lane & 15);
#pragma unroll
    for (int j = 0; j < 8; ++j) {
      const int k = ((j < 4) ? (2 * j) : (8 + 2 * j)) + kh;
      afr[i].u[j] = *(const unsigned*)&Asb[r][k];
    }
  }
  const int ko = (lane >> 4) * 8;          // B: lanes 16..31 hold K+8 group
#pragma unroll
  for (int jn = 0; jn < 4; ++jn) {
    const int c = wn * 64 + jn * 16 + (lane & 15);
#pragma unroll
    for (int j = 0; j < 8; ++j)
      bfr[jn].u[j] = *(const unsigned*)&Bsb[c][((j < 4) ? (2 * j) : (8 + 2 * j)) + ko];
  }
#pragma unroll
  for (int i = 0; i < 2; ++i)
#pragma unroll
    for (int jn = 0; jn < 4; ++jn)
      acc[i][jn] = __builtin_amdgcn_wmma_f32_16x16x32_bf16(
          false, afr[i].v, false, bfr[jn].v, (short)0, acc[i][jn],
          false, false);
}

// =====================================================================
// General batched GEMM:  C = epilogue(alpha * A(f32) x B(f32->bf16) )
// A[M,K] row-major f32 ; B[K,N] (or [N,K] if transB) f32 ; C[M,N] f32.
// Tile 128x128x32, 256 threads = 8 waves; wave tile 32x64 = 2x4 WMMA.
// Fast path: unguarded b128 loads + double-buffered LDS pipeline.
// Edge path: row-guarded (one branch/thread) vectorized staging.
// =====================================================================
struct GemmP {
  const float* A; long long lda, sA0, sA1;
  const float* B; long long ldb, sB0, sB1;
  float*       C; long long ldc, sC0, sC1;
  const float* bias;
  const float* residual; long long ldr;
  const float* rowScale; long long rsStride;
  int M, N, K;
  float alpha;
  int transB, relu, accum, batchDiv;
};

__launch_bounds__(256)
__global__ void gemm_bf16_wmma(GemmP p) {
  __shared__ unsigned short As[2][128][LDK];
  __shared__ unsigned short Bs[2][128][LDK];

  const int tid  = threadIdx.x;
  const int lane = tid & 31;
  const int wave = tid >> 5;
  const int wm   = wave & 3;        // 4 waves along M (32 rows each)
  const int wn   = wave >> 2;       // 2 waves along N (64 cols each)

  const int z  = blockIdx.z;
  const int b0 = z / p.batchDiv, b1 = z % p.batchDiv;
  const float* A  = p.A + (long long)b0 * p.sA0 + (long long)b1 * p.sA1;
  const float* Bm = p.B + (long long)b0 * p.sB0 + (long long)b1 * p.sB1;
  float*       C  = p.C + (long long)b0 * p.sC0 + (long long)b1 * p.sC1;

  const int m0 = blockIdx.y * 128;
  const int n0 = blockIdx.x * 128;

  v8f acc[2][4] = {};

  const int srow = tid >> 1;          // 0..127
  const int skS  = (tid & 1) * 16;    // 0 or 16

  const bool full = (m0 + 128 <= p.M) && (n0 + 128 <= p.N) &&
                    ((p.K & 31) == 0);

  if (full) {
    // ---------------- fast path: pipelined, unguarded ----------------
    float4 aR[4];
    float  bR[16];
    const float* aBase = A + (long long)(m0 + srow) * p.lda + skS;

    auto loadTile = [&](int k0) {
      const float* ap = aBase + k0;
      aR[0] = *(const float4*)(ap);
      aR[1] = *(const float4*)(ap + 4);
      aR[2] = *(const float4*)(ap + 8);
      aR[3] = *(const float4*)(ap + 12);
      if (p.transB) {
        const float* bp = Bm + (long long)(n0 + srow) * p.ldb + k0 + skS;
        const float4 b0v = *(const float4*)(bp);
        const float4 b1v = *(const float4*)(bp + 4);
        const float4 b2v = *(const float4*)(bp + 8);
        const float4 b3v = *(const float4*)(bp + 12);
        bR[0]=b0v.x; bR[1]=b0v.y; bR[2]=b0v.z; bR[3]=b0v.w;
        bR[4]=b1v.x; bR[5]=b1v.y; bR[6]=b1v.z; bR[7]=b1v.w;
        bR[8]=b2v.x; bR[9]=b2v.y; bR[10]=b2v.z; bR[11]=b2v.w;
        bR[12]=b3v.x; bR[13]=b3v.y; bR[14]=b3v.z; bR[15]=b3v.w;
      } else {
        const float* bp = Bm + (long long)(k0 + skS) * p.ldb + (n0 + srow);
#pragma unroll
        for (int j = 0; j < 16; ++j) bR[j] = bp[(long long)j * p.ldb];
      }
    };
    auto storeTile = [&](int buf) {
      uint4* da = (uint4*)&As[buf][srow][skS];
      da[0] = make_uint4(pack2(aR[0].x, aR[0].y), pack2(aR[0].z, aR[0].w),
                         pack2(aR[1].x, aR[1].y), pack2(aR[1].z, aR[1].w));
      da[1] = make_uint4(pack2(aR[2].x, aR[2].y), pack2(aR[2].z, aR[2].w),
                         pack2(aR[3].x, aR[3].y), pack2(aR[3].z, aR[3].w));
      uint4* db = (uint4*)&Bs[buf][srow][skS];
      db[0] = make_uint4(pack2(bR[0], bR[1]),  pack2(bR[2], bR[3]),
                         pack2(bR[4], bR[5]),  pack2(bR[6], bR[7]));
      db[1] = make_uint4(pack2(bR[8], bR[9]),  pack2(bR[10], bR[11]),
                         pack2(bR[12], bR[13]), pack2(bR[14], bR[15]));
    };

    loadTile(0);
    storeTile(0);
    __syncthreads();
    int buf = 0;
    for (int k0 = 0; k0 < p.K; k0 += 32) {
      const int kn = k0 + 32;
      if (kn < p.K) loadTile(kn);               // overlap with WMMA below
      wmma_tile(As[buf], Bs[buf], wm, wn, lane, acc);
      if (kn < p.K) {
        __syncthreads();                        // all waves done with buf^1
        storeTile(buf ^ 1);
        __syncthreads();
        buf ^= 1;
      }
    }

    // unguarded epilogue
#pragma unroll
    for (int i = 0; i < 2; ++i) {
      const int mBase = m0 + wm * 32 + i * 16 + (lane >> 4) * 8;
#pragma unroll
      for (int jn = 0; jn < 4; ++jn) {
        const int nG = n0 + wn * 64 + jn * 16 + (lane & 15);
        const float bia = p.bias ? p.bias[nG] : 0.f;
#pragma unroll
        for (int r = 0; r < 8; ++r) {
          const int m = mBase + r;
          float v = acc[i][jn][r] * p.alpha + bia;
          if (p.relu) v = fmaxf(v, 0.f);
          if (p.rowScale) v *= p.rowScale[(long long)m * p.rsStride];
          if (p.residual) v += p.residual[(long long)m * p.ldr + nG];
          const long long ci = (long long)m * p.ldc + nG;
          if (p.accum) C[ci] += v; else C[ci] = v;
        }
      }
    }
    return;
  }

  // ------- edge path: row-guarded vectorized staging (K%32==0) -------
  const bool ktail = (p.K & 31) != 0;
  const int  gm    = m0 + srow;
  const int  gn    = n0 + srow;
  const bool mok   = gm < p.M;
  const bool nok   = gn < p.N;

  for (int k0 = 0; k0 < p.K; k0 += 32) {
    if (!ktail) {
      // A rows: one branch per thread, b128 loads, zero-fill otherwise
      float4 a0 = make_float4(0.f, 0.f, 0.f, 0.f), a1 = a0, a2 = a0, a3 = a0;
      if (mok) {
        const float* ap = A + (long long)gm * p.lda + k0 + skS;
        a0 = *(const float4*)(ap);
        a1 = *(const float4*)(ap + 4);
        a2 = *(const float4*)(ap + 8);
        a3 = *(const float4*)(ap + 12);
      }
      uint4* da = (uint4*)&As[0][srow][skS];
      da[0] = make_uint4(pack2(a0.x, a0.y), pack2(a0.z, a0.w),
                         pack2(a1.x, a1.y), pack2(a1.z, a1.w));
      da[1] = make_uint4(pack2(a2.x, a2.y), pack2(a2.z, a2.w),
                         pack2(a3.x, a3.y), pack2(a3.z, a3.w));
      // B cols: one branch per thread
      float bv[16];
#pragma unroll
      for (int j = 0; j < 16; ++j) bv[j] = 0.f;
      if (nok) {
        if (p.transB) {
          const float* bp = Bm + (long long)gn * p.ldb + k0 + skS;
          const float4 b0v = *(const float4*)(bp);
          const float4 b1v = *(const float4*)(bp + 4);
          const float4 b2v = *(const float4*)(bp + 8);
          const float4 b3v = *(const float4*)(bp + 12);
          bv[0]=b0v.x; bv[1]=b0v.y; bv[2]=b0v.z; bv[3]=b0v.w;
          bv[4]=b1v.x; bv[5]=b1v.y; bv[6]=b1v.z; bv[7]=b1v.w;
          bv[8]=b2v.x; bv[9]=b2v.y; bv[10]=b2v.z; bv[11]=b2v.w;
          bv[12]=b3v.x; bv[13]=b3v.y; bv[14]=b3v.z; bv[15]=b3v.w;
        } else {
          const float* bp = Bm + (long long)(k0 + skS) * p.ldb + gn;
#pragma unroll
          for (int j = 0; j < 16; ++j) bv[j] = bp[(long long)j * p.ldb];
        }
      }
      uint4* db = (uint4*)&Bs[0][srow][skS];
      db[0] = make_uint4(pack2(bv[0], bv[1]),   pack2(bv[2], bv[3]),
                         pack2(bv[4], bv[5]),   pack2(bv[6], bv[7]));
      db[1] = make_uint4(pack2(bv[8], bv[9]),   pack2(bv[10], bv[11]),
                         pack2(bv[12], bv[13]), pack2(bv[14], bv[15]));
    } else {
      // fully generic K-tail path (not used by this model)
      const float* ap = A + (long long)gm * p.lda + (k0 + skS);
#pragma unroll
      for (int j = 0; j < 16; ++j) {
        const int gk = k0 + skS + j;
        float v = (mok && gk < p.K) ? ap[j] : 0.f;
        As[0][srow][skS + j] = f2bf(v);
      }
#pragma unroll
      for (int j = 0; j < 16; ++j) {
        const int gk = k0 + skS + j;
        float v = 0.f;
        if (nok && gk < p.K)
          v = p.transB ? Bm[(long long)gn * p.ldb + gk]
                       : Bm[(long long)gk * p.ldb + gn];
        Bs[0][srow][skS + j] = f2bf(v);
      }
    }
    __syncthreads();
    wmma_tile(As[0], Bs[0], wm, wn, lane, acc);
    __syncthreads();
  }

#pragma unroll
  for (int i = 0; i < 2; ++i) {
    const int mBase = m0 + wm * 32 + i * 16 + (lane >> 4) * 8;
#pragma unroll
    for (int jn = 0; jn < 4; ++jn) {
      const int nG = n0 + wn * 64 + jn * 16 + (lane & 15);
      if (nG >= p.N) continue;
      const float bia = p.bias ? p.bias[nG] : 0.f;
#pragma unroll
      for (int r = 0; r < 8; ++r) {
        const int m = mBase + r;
        if (m >= p.M) continue;
        float v = acc[i][jn][r] * p.alpha + bia;
        if (p.relu) v = fmaxf(v, 0.f);
        if (p.rowScale) v *= p.rowScale[(long long)m * p.rsStride];
        if (p.residual) v += p.residual[(long long)m * p.ldr + nG];
        const long long ci = (long long)m * p.ldc + nG;
        if (p.accum) C[ci] += v; else C[ci] = v;
      }
    }
  }
}

// ================= embedding + sinusoidal positional encoding =========
__global__ void embed_pe_kernel(const int* __restrict__ tok,
                                const float* __restrict__ emb,
                                float* __restrict__ out, int S, long long total) {
  long long idx = (long long)blockIdx.x * blockDim.x + threadIdx.x;
  if (idx >= total) return;
  const int d   = (int)(idx & (DM - 1));
  const long long row = idx >> 10;        // DM = 1024
  const int s   = (int)(row % S);
  const int t   = tok[row];
  const int i2  = d & ~1;
  const float freq = __expf((float)i2 * (-9.2103403719761836f / (float)DM));
  const float ang  = (float)s * freq;
  const float pe   = (d & 1) ? __cosf(ang) : __sinf(ang);
  out[idx] = emb[(long long)t * DM + d] + pe;
}

// ================= top-2 router (one 128-thread block per token) ======
__global__ void router_kernel(const float* __restrict__ x,
                              const float* __restrict__ wg,
                              const float* __restrict__ bg,
                              float* __restrict__ rw,
                              float* __restrict__ probs) {
  const int row = blockIdx.x;
  const int tid = threadIdx.x;            // 0..127
  float acc[NE];
#pragma unroll
  for (int e = 0; e < NE; ++e) acc[e] = 0.f;
  const float* xr = x + (long long)row * DM;
  for (int d = tid; d < DM; d += 128) {
    const float xv = xr[d];
#pragma unroll
    for (int e = 0; e < NE; ++e) acc[e] += xv * wg[d * NE + e];
  }
  __shared__ float sb[128];
  __shared__ float logit[NE];
  for (int e = 0; e < NE; ++e) {
    sb[tid] = acc[e]; __syncthreads();
    for (int o = 64; o > 0; o >>= 1) { if (tid < o) sb[tid] += sb[tid + o]; __syncthreads(); }
    if (tid == 0) logit[e] = sb[0] + bg[e];
    __syncthreads();
  }
  if (tid == 0) {
    int i1 = 0;
    for (int e = 1; e < NE; ++e) if (logit[e] > logit[i1]) i1 = e;
    int i2 = (i1 == 0) ? 1 : 0;
    for (int e = 0; e < NE; ++e) { if (e == i1) continue; if (logit[e] > logit[i2]) i2 = e; }
    const float mm = fmaxf(logit[i1], logit[i2]);
    const float e1 = __expf(logit[i1] - mm), e2 = __expf(logit[i2] - mm);
    const float s1 = e1 / (e1 + e2),        s2 = e2 / (e1 + e2);
    float mx = logit[0];
    for (int e = 1; e < NE; ++e) mx = fmaxf(mx, logit[e]);
    float den = 0.f, pv[NE];
#pragma unroll
    for (int e = 0; e < NE; ++e) { pv[e] = __expf(logit[e] - mx); den += pv[e]; }
#pragma unroll
    for (int e = 0; e < NE; ++e) {
      probs[(long long)row * NE + e] = pv[e] / den;
      rw[(long long)row * NE + e]    = (e == i1) ? s1 : ((e == i2) ? s2 : 0.f);
    }
  }
}

// ================= deterministic aux-loss reduction ===================
__global__ void aux_kernel(const float* __restrict__ rw,
                           const float* __restrict__ probs,
                           float* __restrict__ out, int T) {
  __shared__ float sb[256];
  const int tid = threadIdx.x;
  float auxv = 0.f;
  for (int e = 0; e < NE; ++e) {
    float imp = 0.f, ld = 0.f;
    for (int t = tid; t < T; t += 256) {
      imp += probs[(long long)t * NE + e];
      if (rw[(long long)t * NE + e] > 0.f) ld += 1.f;
    }
    sb[tid] = imp; __syncthreads();
    for (int o = 128; o > 0; o >>= 1) { if (tid < o) sb[tid] += sb[tid + o]; __syncthreads(); }
    const float impT = sb[0]; __syncthreads();
    sb[tid] = ld; __syncthreads();
    for (int o = 128; o > 0; o >>= 1) { if (tid < o) sb[tid] += sb[tid + o]; __syncthreads(); }
    const float ldT = sb[0]; __syncthreads();
    auxv += impT * (ldT / (float)T) * (float)(NE * NE);
  }
  if (tid == 0) out[0] = auxv / (float)NE;
}

// ================= LayerNorm (block per row, D=1024) ==================
__global__ void layernorm_kernel(const float* __restrict__ x,
                                 const float* __restrict__ g,
                                 const float* __restrict__ b,
                                 float* __restrict__ y) {
  const long long row = blockIdx.x;
  const float* xr = x + row * DM;
  float*       yr = y + row * DM;
  __shared__ float sb[256];
  const int tid = threadIdx.x;
  float s = 0.f;
  for (int d = tid; d < DM; d += 256) s += xr[d];
  sb[tid] = s; __syncthreads();
  for (int o = 128; o > 0; o >>= 1) { if (tid < o) sb[tid] += sb[tid + o]; __syncthreads(); }
  const float mean = sb[0] / (float)DM; __syncthreads();
  float v = 0.f;
  for (int d = tid; d < DM; d += 256) { const float t = xr[d] - mean; v += t * t; }
  sb[tid] = v; __syncthreads();
  for (int o = 128; o > 0; o >>= 1) { if (tid < o) sb[tid] += sb[tid + o]; __syncthreads(); }
  const float rstd = rsqrtf(sb[0] / (float)DM + 1e-5f); __syncthreads();
  for (int d = tid; d < DM; d += 256)
    yr[d] = (xr[d] - mean) * rstd * g[d] + b[d];
}

// ================= row softmax (block per row) ========================
__global__ void softmax_kernel(float* __restrict__ x, int L) {
  const long long row = blockIdx.x;
  float* xr = x + row * (long long)L;
  __shared__ float sb[256];
  const int tid = threadIdx.x;
  float mx = -3.4e38f;
  for (int i = tid; i < L; i += 256) mx = fmaxf(mx, xr[i]);
  sb[tid] = mx; __syncthreads();
  for (int o = 128; o > 0; o >>= 1) { if (tid < o) sb[tid] = fmaxf(sb[tid], sb[tid + o]); __syncthreads(); }
  mx = sb[0]; __syncthreads();
  float s = 0.f;
  for (int i = tid; i < L; i += 256) s += __expf(xr[i] - mx);
  sb[tid] = s; __syncthreads();
  for (int o = 128; o > 0; o >>= 1) { if (tid < o) sb[tid] += sb[tid + o]; __syncthreads(); }
  const float inv = 1.f / sb[0]; __syncthreads();
  for (int i = tid; i < L; i += 256) xr[i] = __expf(xr[i] - mx) * inv;
}

__global__ void zero_kernel(float* __restrict__ p, long long n) {
  long long i = (long long)blockIdx.x * blockDim.x + threadIdx.x;
  if (i < n) p[i] = 0.f;
}

// =====================================================================
extern "C" void kernel_launch(void* const* d_in, const int* in_sizes, int n_in,
                              void* d_out, int out_size, void* d_ws, size_t ws_size,
                              hipStream_t stream) {
  (void)in_sizes; (void)out_size; (void)ws_size;
  if (n_in < 96) return;
  typedef long long ll;

  // ---- input mapping: setup_inputs() dict, recursive insertion order ----
  const int* src = (const int*)d_in[0];
  const int* tgt = (const int*)d_in[1];
  int ix = 2;
  auto nf = [&]() { return (const float*)d_in[ix++]; };
  const float* src_emb = nf();
  const float* tgt_emb = nf();
  const float* wg  = nf();
  const float* bg  = nf();
  const float* ew1 = nf();
  const float* eb1 = nf();
  const float* ew2 = nf();
  const float* eb2 = nf();
  struct Attn { const float *wq,*bq,*wk,*bk,*wv,*bv,*wo,*bo; };
  auto readAttn = [&]() {
    Attn a; a.wq=nf(); a.bq=nf(); a.wk=nf(); a.bk=nf();
            a.wv=nf(); a.bv=nf(); a.wo=nf(); a.bo=nf(); return a;
  };
  struct EncL { Attn a; const float *w1,*b1,*w2,*b2,*l1g,*l1b,*l2g,*l2b; };
  struct DecL { Attn sa, ca; const float *w1,*b1,*w2,*b2,*l1g,*l1b,*l2g,*l2b,*l3g,*l3b; };
  EncL enc[NLAY]; DecL dec[NLAY];
  for (int l = 0; l < NLAY; ++l) {
    enc[l].a = readAttn();
    enc[l].w1=nf(); enc[l].b1=nf(); enc[l].w2=nf(); enc[l].b2=nf();
    enc[l].l1g=nf(); enc[l].l1b=nf(); enc[l].l2g=nf(); enc[l].l2b=nf();
  }
  for (int l = 0; l < NLAY; ++l) {
    dec[l].sa = readAttn(); dec[l].ca = readAttn();
    dec[l].w1=nf(); dec[l].b1=nf(); dec[l].w2=nf(); dec[l].b2=nf();
    dec[l].l1g=nf(); dec[l].l1b=nf(); dec[l].l2g=nf(); dec[l].l2b=nf();
    dec[l].l3g=nf(); dec[l].l3b=nf();
  }
  const float* fcw = nf();
  const float* fcb = nf();

  // ---- workspace carve-up (floats) ----
  float* ws = (float*)d_ws;
  const ll A2M = 2097152;               // B*SS*D
  float* xb   = ws;                     // activations / scratch
  float* yb   = xb + A2M;               // activations / encoder final
  float* ctx  = yb + A2M;               // attention context
  float* qb   = ctx + A2M;
  float* kb   = qb + A2M;
  float* vb   = kb + A2M;
  float* h1   = vb + A2M;               // 2048 x 4096
  float* dx   = h1 + (ll)2048 * 4096;   // decoder activations
  float* rwb  = dx + A2M;               // [2048, E] router weights
  float* prb  = rwb + 2048 * NE;        // [2048, E] full softmax probs
  float* scor = prb + 2048 * NE;        // up to 2*16*1024*1024

  // ---- GEMM launcher ----
  auto G = [&](const float* A, ll lda, ll sA0, ll sA1,
               const float* B, ll ldb, ll sB0, ll sB1, int transB,
               float* C, ll ldc, ll sC0, ll sC1,
               const float* bias, const float* residual, ll ldr,
               const float* rowScale, ll rsStride,
               int M, int N, int K, float alpha, int relu, int accum,
               int batches, int batchDiv) {
    GemmP p;
    p.A=A; p.lda=lda; p.sA0=sA0; p.sA1=sA1;
    p.B=B; p.ldb=ldb; p.sB0=sB0; p.sB1=sB1;
    p.C=C; p.ldc=ldc; p.sC0=sC0; p.sC1=sC1;
    p.bias=bias; p.residual=residual; p.ldr=ldr;
    p.rowScale=rowScale; p.rsStride=rsStride;
    p.M=M; p.N=N; p.K=K; p.alpha=alpha;
    p.transB=transB; p.relu=relu; p.accum=accum; p.batchDiv=batchDiv;
    dim3 grid((N + 127) / 128, (M + 127) / 128, batches);
    gemm_bf16_wmma<<<grid, 256, 0, stream>>>(p);
  };
  auto Glin = [&](const float* A, const float* W, const float* bias, float* C,
                  int M, int N, int K, int relu, const float* residual) {
    G(A, K, 0, 0, W, N, 0, 0, 0, C, N, 0, 0,
      bias, residual, N, nullptr, 0, M, N, K, 1.f, relu, 0, 1, 1);
  };
  auto LN = [&](const float* x, const float* g, const float* b, float* y, int rows) {
    layernorm_kernel<<<rows, 256, 0, stream>>>(x, g, b, y);
  };
  // Full multi-head attention: out = xq + (softmax(QK^T/8) V) Wo + bo
  auto MHA = [&](const float* xq, int Mq, int Sq,
                 const float* xkv, int Mkv, int Skv,
                 const Attn& w, float* outBuf) {
    Glin(xq,  w.wq, w.bq, qb, Mq,  DM, DM, 0, nullptr);
    Glin(xkv, w.wk, w.bk, kb, Mkv, DM, DM, 0, nullptr);
    Glin(xkv, w.wv, w.bv, vb, Mkv, DM, DM, 0, nullptr);
    // scores[b,h,q,k] = (1/8) sum_d Q[b,q,h,d] K[b,k,h,d]
    G(qb, DM, (ll)Sq * DM, DHD,
      kb, DM, (ll)Skv * DM, DHD, 1,
      scor, Skv, (ll)NHH * Sq * Skv, (ll)Sq * Skv,
      nullptr, nullptr, 0, nullptr, 0,
      Sq, Skv, DHD, 0.125f, 0, 0, BBS * NHH, NHH);
    softmax_kernel<<<BBS * NHH * Sq, 256, 0, stream>>>(scor, Skv);
    // ctx[b,q,h,d] = sum_k P[b,h,q,k] V[b,k,h,d]
    G(scor, Skv, (ll)NHH * Sq * Skv, (ll)Sq * Skv,
      vb, DM, (ll)Skv * DM, DHD, 0,
      ctx, DM, (ll)Sq * DM, DHD,
      nullptr, nullptr, 0, nullptr, 0,
      Sq, DHD, Skv, 1.f, 0, 0, BBS * NHH, NHH);
    Glin(ctx, w.wo, w.bo, outBuf, Mq, DM, DM, 0, xq);   // fused residual
  };

  const int M1 = BBS * SSRC;   // 2048 encoder tokens
  const int M2 = BBS * STGT;   // 1024 decoder tokens

  // ---------------- encoder ----------------
  embed_pe_kernel<<<(int)(((ll)M1 * DM + 255) / 256), 256, 0, stream>>>(
      src, src_emb, xb, SSRC, (ll)M1 * DM);
  router_kernel<<<M1, 128, 0, stream>>>(xb, wg, bg, rwb, prb);
  aux_kernel<<<1, 256, 0, stream>>>(rwb, prb,
      (float*)d_out + (ll)M2 * VTGT, M1);

  // dense all-expert MoE: yb = sum_e rw[:,e] * (relu(x ew1_e + b1_e) ew2_e + b2_e)
  zero_kernel<<<(int)(((ll)M1 * DM + 255) / 256), 256, 0, stream>>>(yb, (ll)M1 * DM);
  for (int e = 0; e < NE; ++e) {
    Glin(xb, ew1 + (ll)e * DM * HIDE, eb1 + (ll)e * HIDE, h1, M1, HIDE, DM, 1, nullptr);
    G(h1, HIDE, 0, 0, ew2 + (ll)e * HIDE * DM, DM, 0, 0, 0,
      yb, DM, 0, 0, eb2 + (ll)e * DM, nullptr, 0,
      rwb + e, NE, M1, DM, HIDE, 1.f, 0, /*accum=*/1, 1, 1);
  }

  // encoder layers; activations live in yb, scratch in xb
  for (int l = 0; l < NLAY; ++l) {
    MHA(yb, M1, SSRC, yb, M1, SSRC, enc[l].a, xb);
    LN(xb, enc[l].l1g, enc[l].l1b, yb, M1);
    Glin(yb, enc[l].w1, enc[l].b1, h1, M1, DFFN, DM, 1, nullptr);
    Glin(h1, enc[l].w2, enc[l].b2, xb, M1, DM, DFFN, 0, yb);
    LN(xb, enc[l].l2g, enc[l].l2b, yb, M1);
  }
  // encoder output preserved in yb

  // ---------------- decoder ----------------
  embed_pe_kernel<<<(int)(((ll)M2 * DM + 255) / 256), 256, 0, stream>>>(
      tgt, tgt_emb, dx, STGT, (ll)M2 * DM);
  for (int l = 0; l < NLAY; ++l) {
    MHA(dx, M2, STGT, dx, M2, STGT, dec[l].sa, xb);       // self-attn
    LN(xb, dec[l].l1g, dec[l].l1b, dx, M2);
    MHA(dx, M2, STGT, yb, M1, SSRC, dec[l].ca, xb);       // cross-attn
    LN(xb, dec[l].l2g, dec[l].l2b, dx, M2);
    Glin(dx, dec[l].w1, dec[l].b1, h1, M2, DFFN, DM, 1, nullptr);
    Glin(h1, dec[l].w2, dec[l].b2, xb, M2, DM, DFFN, 0, dx);
    LN(xb, dec[l].l3g, dec[l].l3b, dx, M2);
  }

  // final vocab projection straight into d_out
  Glin(dx, fcw, fcb, (float*)d_out, M2, VTGT, DM, 0, nullptr);
}
